// _MultiheadAttention_4286377361554
// MI455X (gfx1250) — matrix operand order repaired and checked by reading, not compiled, not run on hardware
//
#include <hip/hip_runtime.h>
#include <hip/hip_bf16.h>

// ---------------- problem constants ----------------
#define B_   4
#define L_   4096
#define D_   1024
#define H_   16
#define DK_  64
#define M_TOT (B_ * L_)     // 16384 rows of X
#define EPS_ 1e-5f

typedef __bf16 bf16_t;
typedef __attribute__((ext_vector_type(16))) bf16_t v16bf;
typedef __attribute__((ext_vector_type(8)))  bf16_t v8bf;
typedef __attribute__((ext_vector_type(8)))  float  v8f;

// ---------------- WMMA fragment loaders ----------------
// A fragment (16x32 bf16, row-major source, leading dim ld):
//   lanes 0-15 : M = lane,    K in {0..7, 16..23}
//   lanes 16-31: M = lane-16, K in {8..15, 24..31}
__device__ __forceinline__ v16bf load_a_frag(const bf16_t* __restrict__ base,
                                             int ld, int row0, int k0, int lane) {
  int m  = lane & 15;
  int kb = (lane >> 4) ? 8 : 0;
  const bf16_t* p = base + (size_t)(row0 + m) * (size_t)ld + (size_t)(k0 + kb);
  v8bf lo = *(const v8bf*)p;          // K = k0+kb   .. +7
  v8bf hi = *(const v8bf*)(p + 16);   // K = k0+16+kb.. +7
  v16bf r;
#pragma unroll
  for (int i = 0; i < 8; ++i) { r[i] = lo[i]; r[i + 8] = hi[i]; }
  return r;
}

// B fragment (32x16 bf16) loaded from B^T storage (BT[n][k], contiguous in k):
//   lanes 0-15 : N = lane,    K in {0..7, 16..23}
//   lanes 16-31: N = lane-16, K in {8..15, 24..31}
__device__ __forceinline__ v16bf load_bt_frag(const bf16_t* __restrict__ bt,
                                              int ld, int n0, int k0, int lane) {
  int n  = lane & 15;
  int kb = (lane >> 4) ? 8 : 0;
  const bf16_t* p = bt + (size_t)(n0 + n) * (size_t)ld + (size_t)(k0 + kb);
  v8bf lo = *(const v8bf*)p;
  v8bf hi = *(const v8bf*)(p + 16);
  v16bf r;
#pragma unroll
  for (int i = 0; i < 8; ++i) { r[i] = lo[i]; r[i + 8] = hi[i]; }
  return r;
}

__device__ __forceinline__ v8f wmma_bf16(v16bf a, v16bf b, v8f c) {
  return __builtin_amdgcn_wmma_f32_16x16x32_bf16(false, a, false, b,
                                                 (short)0, c, false, false);
}

__device__ __forceinline__ float phi_elu1(float x) {
  return x > 0.0f ? x + 1.0f : __expf(x);   // elu(x)+1
}

// ---------------- kernel 0: f32 -> bf16 convert ----------------
__global__ __launch_bounds__(256) void cvt_bf16_kernel(const float* __restrict__ src,
                                                       bf16_t* __restrict__ dst, int n) {
  int i = blockIdx.x * 256 + threadIdx.x;
  if (i < n) dst[i] = (bf16_t)src[i];
}

// ---------------- kernel 1: projection GEMM  out = phi?(X @ W^T + b) ----------------
// MODE 0: phi, store row-major bf16 [M_TOT, D]            (phi_q)
// MODE 1: phi, store transposed bf16 [B,H,DK,L]           (phi_k^T)
// MODE 2: no phi, store transposed bf16 [B,H,DK,L]        (v^T)
// Block tile 128x128, 8 waves (256 thr), wave tile 32(M) x 64(N), K step 32,
// register double-buffered so next-k loads overlap current WMMAs.
template <int MODE>
__global__ __launch_bounds__(256) void proj_kernel(const bf16_t* __restrict__ X,
                                                   const bf16_t* __restrict__ W,
                                                   const float* __restrict__ bias,
                                                   bf16_t* __restrict__ out) {
  const int tid  = threadIdx.x;
  const int lane = tid & 31;
  const int w    = tid >> 5;
  const int wm   = w >> 1;            // 0..3
  const int wn   = w & 1;             // 0..1
  const int row0 = blockIdx.y * 128 + wm * 32;
  const int col0 = blockIdx.x * 128 + wn * 64;

  v8f acc[2][4] = {};

  // prologue loads (k0 = 0)
  v16bf a0 = load_a_frag(X, D_, row0,      0, lane);
  v16bf a1 = load_a_frag(X, D_, row0 + 16, 0, lane);
  v16bf b[4];
#pragma unroll
  for (int ni = 0; ni < 4; ++ni) b[ni] = load_bt_frag(W, D_, col0 + 16 * ni, 0, lane);

  for (int k0 = 32; k0 < D_; k0 += 32) {
    v16bf a0n = load_a_frag(X, D_, row0,      k0, lane);
    v16bf a1n = load_a_frag(X, D_, row0 + 16, k0, lane);
    v16bf bn[4];
#pragma unroll
    for (int ni = 0; ni < 4; ++ni) bn[ni] = load_bt_frag(W, D_, col0 + 16 * ni, k0, lane);
#pragma unroll
    for (int ni = 0; ni < 4; ++ni) {
      acc[0][ni] = wmma_bf16(a0, b[ni], acc[0][ni]);
      acc[1][ni] = wmma_bf16(a1, b[ni], acc[1][ni]);
    }
    a0 = a0n; a1 = a1n;
#pragma unroll
    for (int ni = 0; ni < 4; ++ni) b[ni] = bn[ni];
  }
#pragma unroll
  for (int ni = 0; ni < 4; ++ni) {
    acc[0][ni] = wmma_bf16(a0, b[ni], acc[0][ni]);
    acc[1][ni] = wmma_bf16(a1, b[ni], acc[1][ni]);
  }

  float bv[4];
#pragma unroll
  for (int ni = 0; ni < 4; ++ni) bv[ni] = bias[col0 + 16 * ni + (lane & 15)];

#pragma unroll
  for (int mi = 0; mi < 2; ++mi) {
#pragma unroll
    for (int ni = 0; ni < 4; ++ni) {
      if (MODE == 0) {
        // row-major scatter store (one bf16 per accumulator element)
        int col = col0 + 16 * ni + (lane & 15);
#pragma unroll
        for (int r = 0; r < 8; ++r) {
          int row = row0 + 16 * mi + r + 8 * (lane >> 4);
          float x = acc[mi][ni][r] + bv[ni];
          x = phi_elu1(x);
          out[(size_t)row * D_ + col] = (bf16_t)x;
        }
      } else {
        // transposed store: contiguous 8 rows (l) per lane -> one b128 store
        int col   = col0 + 16 * ni + (lane & 15);     // global feature index d
        int h     = col >> 6;
        int dk    = col & 63;
        int rbase = row0 + 16 * mi + 8 * (lane >> 4); // 8 consecutive global rows
        int b_    = rbase >> 12;                      // / L_
        int l     = rbase & (L_ - 1);
        v8bf pack;
#pragma unroll
        for (int r = 0; r < 8; ++r) {
          float x = acc[mi][ni][r] + bv[ni];
          if (MODE == 1) x = phi_elu1(x);
          pack[r] = (bf16_t)x;
        }
        size_t idx = (((size_t)(b_ * H_ + h) * DK_ + dk) * L_ + l);
        *(v8bf*)(out + idx) = pack;
      }
    }
  }
}

// ---------------- kernel 2: KV = phi_k^T @ v  (per b,h: 64x64xK=4096) + K_sum ----------------
// 64 blocks (one per b,h), 512 threads = 16 waves; wave (mt,nt) owns one 16x16 tile.
// K_sum is computed on the matrix unit: extra WMMA against an all-ones B fragment
// (every accumulator column = row sum) -> zero VALU work in the hot loop.
// Stores KV transposed (KVT[m][d], bf16) for contiguous B loads in kernel 3.
__global__ __launch_bounds__(512) void kv_kernel(const bf16_t* __restrict__ phi_kT,
                                                 const bf16_t* __restrict__ vT,
                                                 bf16_t* __restrict__ KVT,
                                                 float* __restrict__ Ksum) {
  const int bh   = blockIdx.x;
  const int tid  = threadIdx.x;
  const int lane = tid & 31;
  const int w    = tid >> 5;
  const int mt   = w >> 2;    // d-tile 0..3
  const int nt   = w & 3;     // m-tile 0..3
  const bf16_t* A  = phi_kT + (size_t)bh * DK_ * L_;  // A[d][l]
  const bf16_t* Bt = vT     + (size_t)bh * DK_ * L_;  // BT[m][l]

  v16bf ones;
#pragma unroll
  for (int i = 0; i < 16; ++i) ones[i] = (bf16_t)1.0f;

  v8f acc = {};
  v8f ksacc = {};
  const int row0 = mt * 16, col0 = nt * 16;

  v16bf a = load_a_frag(A, L_, row0, 0, lane);
  v16bf b = load_bt_frag(Bt, L_, col0, 0, lane);
  for (int k0 = 32; k0 < L_; k0 += 32) {
    v16bf an = load_a_frag(A, L_, row0, k0, lane);
    v16bf bn = load_bt_frag(Bt, L_, col0, k0, lane);
    acc   = wmma_bf16(a, b,    acc);
    ksacc = wmma_bf16(a, ones, ksacc);
    a = an; b = bn;
  }
  acc   = wmma_bf16(a, b,    acc);
  ksacc = wmma_bf16(a, ones, ksacc);

  // K_sum store: column 0 of ksacc (lanes 0 and 16) holds 8 consecutive d each.
  if (nt == 0 && (lane & 15) == 0) {
    float* dst = Ksum + bh * 64 + row0 + 8 * (lane >> 4);
#pragma unroll
    for (int r = 0; r < 8; ++r) dst[r] = ksacc[r];
  }

  // KVT[m][d] store: 8 consecutive d per lane -> one b128 store
  int m     = col0 + (lane & 15);
  int dbase = row0 + 8 * (lane >> 4);
  v8bf pack;
#pragma unroll
  for (int r = 0; r < 8; ++r) pack[r] = (bf16_t)acc[r];
  *(v8bf*)(KVT + (size_t)bh * DK_ * DK_ + (size_t)m * DK_ + dbase) = pack;
}

// ---------------- kernel 3: V_new = (phi_q @ KV) * Z ----------------
// grid (L/128, B*H), 256 threads = 8 waves; wave owns 16(M) x 64(N), K=64.
__global__ __launch_bounds__(256) void attn_kernel(const bf16_t* __restrict__ phi_q,
                                                   const bf16_t* __restrict__ KVT,
                                                   const float* __restrict__ Ksum,
                                                   bf16_t* __restrict__ Vnew) {
  const int bh   = blockIdx.y;
  const int b    = bh >> 4, h = bh & 15;
  const int tid  = threadIdx.x;
  const int lane = tid & 31;
  const int w    = tid >> 5;
  const int l0   = blockIdx.x * 128 + w * 16;
  const int arow0 = b * L_ + l0;

  const bf16_t* Abase = phi_q + h * 64;               // column offset into head
  const bf16_t* Bt    = KVT + (size_t)bh * DK_ * DK_; // BT[m][d]
  const float*  ks    = Ksum + bh * 64;

  __shared__ float zp[8][32];
  __shared__ float z16[8][16];

  v8f acc[4] = {};
#pragma unroll
  for (int k0 = 0; k0 < DK_; k0 += 32) {
    v16bf a = load_a_frag(Abase, D_, arow0, k0, lane);
#pragma unroll
    for (int ni = 0; ni < 4; ++ni) {
      v16bf bb = load_bt_frag(Bt, DK_, ni * 16, k0, lane);
      acc[ni] = wmma_bf16(a, bb, acc[ni]);
    }
  }

  // Z[l] = 1 / (phi_q[l] . K_sum + eps); 2 lanes per row, 32 elems each,
  // 4 independent accumulation chains.
  {
    int r = lane >> 1, half = lane & 1;
    const bf16_t* prow = phi_q + (size_t)(arow0 + r) * D_ + h * 64 + half * 32;
    const float4* k4   = (const float4*)(ks + half * 32);
    float parts[4] = {0.f, 0.f, 0.f, 0.f};
#pragma unroll
    for (int c = 0; c < 4; ++c) {
      v8bf pv = *(const v8bf*)(prow + 8 * c);
      float4 ka = k4[2 * c], kb_ = k4[2 * c + 1];
      parts[c] += (float)pv[0] * ka.x;  parts[c] += (float)pv[1] * ka.y;
      parts[c] += (float)pv[2] * ka.z;  parts[c] += (float)pv[3] * ka.w;
      parts[c] += (float)pv[4] * kb_.x; parts[c] += (float)pv[5] * kb_.y;
      parts[c] += (float)pv[6] * kb_.z; parts[c] += (float)pv[7] * kb_.w;
    }
    zp[w][lane] = (parts[0] + parts[1]) + (parts[2] + parts[3]);
  }
  __syncthreads();
  if (lane < 16)
    z16[w][lane] = 1.0f / (zp[w][2 * lane] + zp[w][2 * lane + 1] + EPS_);
  __syncthreads();

  // scale + row-major bf16 store of V_new
#pragma unroll
  for (int ni = 0; ni < 4; ++ni) {
    int col = h * 64 + ni * 16 + (lane & 15);
#pragma unroll
    for (int r = 0; r < 8; ++r) {
      int ml  = r + 8 * (lane >> 4);
      int row = arow0 + ml;
      float val = acc[ni][r] * z16[w][ml];
      Vnew[(size_t)row * D_ + col] = (bf16_t)val;
    }
  }
}

// ---------------- kernel 4: out = V_new @ W_O^T + b_O (f32) ----------------
__global__ __launch_bounds__(256) void out_kernel(const bf16_t* __restrict__ Vn,
                                                  const bf16_t* __restrict__ W,
                                                  const float* __restrict__ bias,
                                                  float* __restrict__ out) {
  const int tid  = threadIdx.x;
  const int lane = tid & 31;
  const int w    = tid >> 5;
  const int wm   = w >> 1, wn = w & 1;
  const int row0 = blockIdx.y * 128 + wm * 32;
  const int col0 = blockIdx.x * 128 + wn * 64;

  v8f acc[2][4] = {};

  v16bf a0 = load_a_frag(Vn, D_, row0,      0, lane);
  v16bf a1 = load_a_frag(Vn, D_, row0 + 16, 0, lane);
  v16bf b[4];
#pragma unroll
  for (int ni = 0; ni < 4; ++ni) b[ni] = load_bt_frag(W, D_, col0 + 16 * ni, 0, lane);

  for (int k0 = 32; k0 < D_; k0 += 32) {
    v16bf a0n = load_a_frag(Vn, D_, row0,      k0, lane);
    v16bf a1n = load_a_frag(Vn, D_, row0 + 16, k0, lane);
    v16bf bn[4];
#pragma unroll
    for (int ni = 0; ni < 4; ++ni) bn[ni] = load_bt_frag(W, D_, col0 + 16 * ni, k0, lane);
#pragma unroll
    for (int ni = 0; ni < 4; ++ni) {
      acc[0][ni] = wmma_bf16(a0, b[ni], acc[0][ni]);
      acc[1][ni] = wmma_bf16(a1, b[ni], acc[1][ni]);
    }
    a0 = a0n; a1 = a1n;
#pragma unroll
    for (int ni = 0; ni < 4; ++ni) b[ni] = bn[ni];
  }
#pragma unroll
  for (int ni = 0; ni < 4; ++ni) {
    acc[0][ni] = wmma_bf16(a0, b[ni], acc[0][ni]);
    acc[1][ni] = wmma_bf16(a1, b[ni], acc[1][ni]);
  }

#pragma unroll
  for (int ni = 0; ni < 4; ++ni) {
    int col = col0 + 16 * ni + (lane & 15);
    float bv = bias[col];
#pragma unroll
    for (int mi = 0; mi < 2; ++mi)
#pragma unroll
      for (int r = 0; r < 8; ++r) {
        int row = row0 + 16 * mi + r + 8 * (lane >> 4);
        out[(size_t)row * D_ + col] = acc[mi][ni][r] + bv;
      }
  }
}

// ---------------- host launcher ----------------
extern "C" void kernel_launch(void* const* d_in, const int* in_sizes, int n_in,
                              void* d_out, int out_size, void* d_ws, size_t ws_size,
                              hipStream_t stream) {
  (void)in_sizes; (void)n_in; (void)out_size; (void)ws_size;
  const float* Qf  = (const float*)d_in[0];
  const float* WQf = (const float*)d_in[1];
  const float* bQ  = (const float*)d_in[2];
  const float* WKf = (const float*)d_in[3];
  const float* bK  = (const float*)d_in[4];
  const float* WVf = (const float*)d_in[5];
  const float* bV  = (const float*)d_in[6];
  const float* WOf = (const float*)d_in[7];
  const float* bO  = (const float*)d_in[8];
  float* outp = (float*)d_out;

  // workspace layout (bf16 element counts); X region is reused for V_new.
  const size_t SZ_X = (size_t)M_TOT * D_;   // 16.7M elems
  const size_t SZ_W = (size_t)D_ * D_;      // 1M elems
  bf16_t* Xb    = (bf16_t*)d_ws;            // X bf16, later V_new bf16
  bf16_t* Wqb   = Xb + SZ_X;
  bf16_t* Wkb   = Wqb + SZ_W;
  bf16_t* Wvb   = Wkb + SZ_W;
  bf16_t* Wob   = Wvb + SZ_W;
  bf16_t* PhiQ  = Wob + SZ_W;               // [M_TOT, D] row-major
  bf16_t* PhiKT = PhiQ + SZ_X;              // [B,H,DK,L]
  bf16_t* VT    = PhiKT + SZ_X;             // [B,H,DK,L]
  bf16_t* KVT   = VT + SZ_X;                // [B*H, 64, 64]  (KVT[m][d])
  float*  Ksum  = (float*)(KVT + (size_t)B_ * H_ * DK_ * DK_); // [B*H, 64]

  // 0) f32 -> bf16 conversions
  cvt_bf16_kernel<<<(int)(SZ_X / 256), 256, 0, stream>>>(Qf,  Xb,  (int)SZ_X);
  cvt_bf16_kernel<<<(int)(SZ_W / 256), 256, 0, stream>>>(WQf, Wqb, (int)SZ_W);
  cvt_bf16_kernel<<<(int)(SZ_W / 256), 256, 0, stream>>>(WKf, Wkb, (int)SZ_W);
  cvt_bf16_kernel<<<(int)(SZ_W / 256), 256, 0, stream>>>(WVf, Wvb, (int)SZ_W);
  cvt_bf16_kernel<<<(int)(SZ_W / 256), 256, 0, stream>>>(WOf, Wob, (int)SZ_W);

  // 1) projections (WMMA): phi_q, phi_k^T, v^T
  dim3 pgrid(D_ / 128, M_TOT / 128);
  proj_kernel<0><<<pgrid, 256, 0, stream>>>(Xb, Wqb, bQ, PhiQ);
  proj_kernel<1><<<pgrid, 256, 0, stream>>>(Xb, Wkb, bK, PhiKT);
  proj_kernel<2><<<pgrid, 256, 0, stream>>>(Xb, Wvb, bV, VT);

  // 2) KV = phi_k^T @ v (per head) + K_sum (via ones-WMMA)
  kv_kernel<<<B_ * H_, 512, 0, stream>>>(PhiKT, VT, KVT, Ksum);

  // 3) V_new = (phi_q @ KV) * Z   (reuses X buffer)
  dim3 agrid(L_ / 128, B_ * H_);
  attn_kernel<<<agrid, 256, 0, stream>>>(PhiQ, KVT, Ksum, Xb);

  // 4) out = V_new @ W_O^T + b_O
  out_kernel<<<pgrid, 256, 0, stream>>>(Xb, Wob, bO, outp);
}
